// MultiHeadSelfAttentionModule_36369783062767
// MI455X (gfx1250) — matrix operand order repaired and checked by reading, not compiled, not run on hardware
//
#include <hip/hip_runtime.h>
#include <hip/hip_bf16.h>
#include <math.h>

// ---------------------------------------------------------------------------
// MHSA block for MI455X (gfx1250, wave32).
// bf16 WMMA (v_wmma_f32_16x16x32_bf16) for all GEMMs, fp32 accumulation.
// Flash-style attention (online softmax) avoids materializing [H,B,S,S] probs.
// GEMM K-loops are software-pipelined with a 2-deep *alternating* buffer set
// (no rotating copies -> no v_mov tax, partial s_wait_loadcnt overlap).
// ---------------------------------------------------------------------------

typedef __bf16 bf16_t;
typedef __attribute__((ext_vector_type(16))) __bf16 v16bf;
typedef __attribute__((ext_vector_type(8)))  __bf16 v8bf;
typedef __attribute__((ext_vector_type(8)))  float  v8f;

#define NB   8       // batch
#define SS   1024    // sequence
#define MM   1024    // model dim
#define HH   16      // heads
#define KDIM 64
#define VDIM 64
#define NQK  2048    // qk projection cols
#define NQKV 3072    // qk + v projection cols

__device__ __forceinline__ v16bf cat8(v8bf lo, v8bf hi) {
  return __builtin_shufflevector(lo, hi, 0,1,2,3,4,5,6,7,8,9,10,11,12,13,14,15);
}
__device__ __forceinline__ v8f wmma_bf16(v16bf a, v16bf b, v8f c) {
  return __builtin_amdgcn_wmma_f32_16x16x32_bf16(false, a, false, b, (short)0, c, false, false);
}
// A-fragment (16x32 bf16, row-major source). ISA layout: lane holds row=lane%16;
// elems 0..7 -> K = 8*half+0..7 ; elems 8..15 -> K = 16+8*half+0..7  (half=lane/16)
__device__ __forceinline__ v16bf load_a_frag(const bf16_t* row, int kbase, int half) {
  v8bf lo = *(const v8bf*)(row + kbase + 8 * half);
  v8bf hi = *(const v8bf*)(row + kbase + 16 + 8 * half);
  return cat8(lo, hi);
}
// B-fragment (32x16 bf16) from [N,K]-stored source: lane holds col N=lane%16;
// elems 0..7 -> K = 8*half+0..7 ; elems 8..15 -> K = 16+8*half+0..7 (two b128 loads)
__device__ __forceinline__ v16bf load_b_frag(const bf16_t* colRow, int kbase, int half) {
  v8bf lo = *(const v8bf*)(colRow + kbase + 8 * half);
  v8bf hi = *(const v8bf*)(colRow + kbase + 16 + 8 * half);
  return cat8(lo, hi);
}

// ------------------------- prep: fp32 -> bf16 staging -----------------------
__global__ __launch_bounds__(256) void cvt_x_kernel(const float* __restrict__ x,
                                                    bf16_t* __restrict__ xbf) {
  size_t i = (size_t)blockIdx.x * 256 + threadIdx.x;   // NB*SS*MM elements
  xbf[i] = (bf16_t)x[i];
}
// wqkv_t[n][k]: n<2048 -> w_qk[k][n], else w_v[k][n-2048]   ([N,K] transposed bf16)
__global__ __launch_bounds__(256) void build_wqkv_kernel(const float* __restrict__ w_qk,
                                                         const float* __restrict__ w_v,
                                                         bf16_t* __restrict__ wt) {
  size_t i = (size_t)blockIdx.x * 256 + threadIdx.x;   // NQKV*MM
  int n = (int)(i / MM), k = (int)(i % MM);
  float v = (n < NQK) ? w_qk[(size_t)k * NQK + n] : w_v[(size_t)k * MM + (n - NQK)];
  wt[i] = (bf16_t)v;
}
__global__ __launch_bounds__(256) void build_wo_kernel(const float* __restrict__ w_o,
                                                       bf16_t* __restrict__ wt) {
  size_t i = (size_t)blockIdx.x * 256 + threadIdx.x;   // MM*MM
  int n = (int)(i / MM), k = (int)(i % MM);
  wt[i] = (bf16_t)w_o[(size_t)k * MM + n];
}

// 2-deep alternating-buffer pipelined GEMM core: acc[0..3] += A(16xK) * B(Kx64)
__device__ __forceinline__ void gemm_16x64_pipelined(const bf16_t* __restrict__ arow,
                                                     const bf16_t* const* wrow,
                                                     int half, v8f acc[4]) {
  v16bf a0 = load_a_frag(arow, 0, half);
  v16bf a1 = load_a_frag(arow, 32, half);
  v16bf b0[4], b1[4];
#pragma unroll
  for (int t = 0; t < 4; ++t) b0[t] = load_b_frag(wrow[t], 0, half);
#pragma unroll
  for (int t = 0; t < 4; ++t) b1[t] = load_b_frag(wrow[t], 32, half);
  for (int ks = 0; ks < MM / 32 - 2; ks += 2) {
    // stage ks: consume buf0, then refill buf0 with stage ks+2
#pragma unroll
    for (int t = 0; t < 4; ++t) acc[t] = wmma_bf16(a0, b0[t], acc[t]);
    a0 = load_a_frag(arow, (ks + 2) * 32, half);
#pragma unroll
    for (int t = 0; t < 4; ++t) b0[t] = load_b_frag(wrow[t], (ks + 2) * 32, half);
    // stage ks+1: consume buf1, then refill buf1 with stage ks+3
#pragma unroll
    for (int t = 0; t < 4; ++t) acc[t] = wmma_bf16(a1, b1[t], acc[t]);
    a1 = load_a_frag(arow, (ks + 3) * 32, half);
#pragma unroll
    for (int t = 0; t < 4; ++t) b1[t] = load_b_frag(wrow[t], (ks + 3) * 32, half);
  }
#pragma unroll
  for (int t = 0; t < 4; ++t) acc[t] = wmma_bf16(a0, b0[t], acc[t]);
#pragma unroll
  for (int t = 0; t < 4; ++t) acc[t] = wmma_bf16(a1, b1[t], acc[t]);
}

// ------------------------- fused QKV projection -----------------------------
// C[(b,s), n] = x @ [w_qk | w_v] + bias ; scatter into q[h][b][s][kd],
// k[h][b][s][kd], v_t[h][b][vd][s] (bf16, laid out for attention fragments).
__global__ __launch_bounds__(256) void qkv_kernel(const bf16_t* __restrict__ xbf,
                                                  const bf16_t* __restrict__ wt,
                                                  const float* __restrict__ b_qk,
                                                  const float* __restrict__ b_v,
                                                  bf16_t* __restrict__ q,
                                                  bf16_t* __restrict__ k,
                                                  bf16_t* __restrict__ vt) {
  const int wave = threadIdx.x >> 5, lane = threadIdx.x & 31;
  const int half = lane >> 4, l16 = lane & 15;
  const int task = blockIdx.x * 8 + wave;          // 512 row-tiles x 48 col-groups
  const int rt = task / 48, cg = task % 48;
  const int row0 = rt * 16, c0 = cg * 64;

  const bf16_t* arow = xbf + (size_t)(row0 + l16) * MM;
  const bf16_t* wrow[4];
#pragma unroll
  for (int t = 0; t < 4; ++t) wrow[t] = wt + (size_t)(c0 + t * 16 + l16) * MM;

  v8f acc[4] = {};
  gemm_16x64_pipelined(arow, wrow, half, acc);

#pragma unroll
  for (int t = 0; t < 4; ++t) {
    const int c = c0 + t * 16 + l16;
    const float bias = (c < NQK) ? b_qk[c] : b_v[c - NQK];
#pragma unroll
    for (int r = 0; r < 8; ++r) {
      const int row = row0 + r + 8 * half;
      const int b = row >> 10, s = row & (SS - 1);
      const bf16_t val = (bf16_t)(acc[t][r] + bias);
      if (c < NQK) {  // col = kd*32 + n*16 + h
        const int kd = c >> 5, n = (c >> 4) & 1, h = c & 15;
        bf16_t* dst = n ? k : q;
        dst[(((size_t)h * NB + b) * SS + s) * KDIM + kd] = val;
      } else {        // col-2048 = vd*16 + h ; store V transposed [h][b][vd][s]
        const int cv = c - NQK, vd = cv >> 4, h = cv & 15;
        vt[(((size_t)h * NB + b) * VDIM + vd) * SS + s] = val;
      }
    }
  }
}

// ------------------------- flash attention ---------------------------------
// One wave per (h, b, 16-row query tile): streams 32-wide key/value slabs,
// online softmax in fp32, P bounced through LDS to repack C-layout -> A-frag.
// V fragments are loaded right after the score WMMAs so the softmax VALU and
// LDS bounce hide their latency.
__global__ __launch_bounds__(256) void attn_kernel(const bf16_t* __restrict__ q,
                                                   const bf16_t* __restrict__ k,
                                                   const bf16_t* __restrict__ vt,
                                                   const unsigned char* __restrict__ mask,
                                                   bf16_t* __restrict__ obf) {
  __shared__ __align__(16) bf16_t pbuf[8][16 * 40];   // 16 rows x 32 cols, padded
  const int wave = threadIdx.x >> 5, lane = threadIdx.x & 31;
  const int half = lane >> 4, l16 = lane & 15;
  const int task = blockIdx.x * 8 + wave;  // 8192 = HH*NB*(SS/16)
  const int it = task & 63;
  const int hb = task >> 6;
  const int h = hb >> 3, b = hb & 7;

  const bf16_t* qbase = q  + ((size_t)h * NB + b) * SS * KDIM;
  const bf16_t* kbase = k  + ((size_t)h * NB + b) * SS * KDIM;
  const bf16_t* vbase = vt + ((size_t)h * NB + b) * VDIM * SS;
  const unsigned char* mrow = mask + (size_t)b * SS;

  // Q fragments for this query tile (kd = 0..31, 32..63), kept in registers
  const bf16_t* qrow = qbase + (size_t)(it * 16 + l16) * KDIM;
  v16bf qa0 = load_a_frag(qrow, 0, half);
  v16bf qa1 = load_a_frag(qrow, 32, half);

  v8f oacc[4] = {};
  float mr[8], lr[8];
#pragma unroll
  for (int r = 0; r < 8; ++r) { mr[r] = -1e30f; lr[r] = 0.f; }
  const float scale = 0.125f;  // 1/sqrt(64)
  bf16_t* myp = &pbuf[wave][0];

  for (int jp = 0; jp < SS / 32; ++jp) {
    // ---- scores for two 16-wide key tiles
    v8f sc[2];
#pragma unroll
    for (int sub = 0; sub < 2; ++sub) {
      const int j0 = jp * 32 + sub * 16;
      const bf16_t* krow = kbase + (size_t)(j0 + l16) * KDIM;
      v8f s = {};
      s = wmma_bf16(qa0, load_b_frag(krow, 0, half), s);
      s = wmma_bf16(qa1, load_b_frag(krow, 32, half), s);
      const bool msk = mrow[j0 + l16] != 0;   // mask is per key column
#pragma unroll
      for (int r = 0; r < 8; ++r) s[r] = msk ? -1e30f : s[r] * scale;
      sc[sub] = s;
    }
    // ---- prefetch PV B-fragments (independent of softmax): latency hidden
    v16bf vb[4];
#pragma unroll
    for (int t = 0; t < 4; ++t) {
      const bf16_t* vrow = vbase + (size_t)(t * 16 + l16) * SS + jp * 32;
      vb[t] = load_b_frag(vrow, 0, half);
    }
    // ---- online softmax (fp32)
    float tmax[8];
#pragma unroll
    for (int r = 0; r < 8; ++r) tmax[r] = fmaxf(sc[0][r], sc[1][r]);
#pragma unroll
    for (int off = 1; off < 16; off <<= 1)
#pragma unroll
      for (int r = 0; r < 8; ++r) tmax[r] = fmaxf(tmax[r], __shfl_xor(tmax[r], off, 32));

    float corr[8], rsum[8];
#pragma unroll
    for (int r = 0; r < 8; ++r) {
      const float mn = fmaxf(mr[r], tmax[r]);
      corr[r] = __expf(mr[r] - mn);
      mr[r] = mn;
      const float p0 = __expf(sc[0][r] - mn);
      const float p1 = __expf(sc[1][r] - mn);
      sc[0][r] = p0; sc[1][r] = p1;
      rsum[r] = p0 + p1;
    }
#pragma unroll
    for (int off = 1; off < 16; off <<= 1)
#pragma unroll
      for (int r = 0; r < 8; ++r) rsum[r] += __shfl_xor(rsum[r], off, 32);
#pragma unroll
    for (int r = 0; r < 8; ++r) lr[r] = lr[r] * corr[r] + rsum[r];
#pragma unroll
    for (int t = 0; t < 4; ++t)
#pragma unroll
      for (int r = 0; r < 8; ++r) oacc[t][r] *= corr[r];

    // ---- repack P: C-layout (row=r+8*half, col=l16) -> LDS -> A-fragment
#pragma unroll
    for (int sub = 0; sub < 2; ++sub)
#pragma unroll
      for (int r = 0; r < 8; ++r)
        myp[(r + 8 * half) * 40 + sub * 16 + l16] = (bf16_t)sc[sub][r];
    asm volatile("s_wait_dscnt 0" ::: "memory");
    v16bf pa = load_a_frag(myp + (size_t)l16 * 40, 0, half);

#pragma unroll
    for (int t = 0; t < 4; ++t) oacc[t] = wmma_bf16(pa, vb[t], oacc[t]);
  }
  // normalize and scatter: obf[(b,s), h*64 + vd]
#pragma unroll
  for (int t = 0; t < 4; ++t)
#pragma unroll
    for (int r = 0; r < 8; ++r) {
      const int s = it * 16 + r + 8 * half;
      obf[((size_t)b * SS + s) * MM + h * VDIM + t * 16 + l16] =
          (bf16_t)(oacc[t][r] / lr[r]);
    }
}

// ------------------- output projection + residual + LN partials ------------
__global__ __launch_bounds__(256) void oproj_kernel(const bf16_t* __restrict__ obf,
                                                    const bf16_t* __restrict__ wot,
                                                    const float* __restrict__ b_o,
                                                    const float* __restrict__ x,
                                                    float* __restrict__ res,
                                                    float* __restrict__ partials) {
  const int wave = threadIdx.x >> 5, lane = threadIdx.x & 31;
  const int half = lane >> 4, l16 = lane & 15;
  const int task = blockIdx.x * 8 + wave;          // 512 row-tiles x 16 col-groups
  const int rt = task / 16, cg = task % 16;
  const int row0 = rt * 16, c0 = cg * 64;

  const bf16_t* arow = obf + (size_t)(row0 + l16) * MM;
  const bf16_t* wrow[4];
#pragma unroll
  for (int t = 0; t < 4; ++t) wrow[t] = wot + (size_t)(c0 + t * 16 + l16) * MM;

  v8f acc[4] = {};
  gemm_16x64_pipelined(arow, wrow, half, acc);

  float sum = 0.f, sumsq = 0.f;
#pragma unroll
  for (int t = 0; t < 4; ++t) {
    const int c = c0 + t * 16 + l16;
    const float bias = b_o[c];
#pragma unroll
    for (int r = 0; r < 8; ++r) {
      const int row = row0 + r + 8 * half;
      const float v = acc[t][r] + bias + x[(size_t)row * MM + c];
      res[(size_t)row * MM + c] = v;
      sum += v; sumsq += v * v;
    }
  }
#pragma unroll
  for (int off = 1; off < 32; off <<= 1) {
    sum   += __shfl_xor(sum,   off, 32);
    sumsq += __shfl_xor(sumsq, off, 32);
  }
  if (lane == 0) { partials[2 * task] = sum; partials[2 * task + 1] = sumsq; }
}

// ------------------- per-batch LN stats (deterministic) ---------------------
__global__ __launch_bounds__(256) void ln_reduce_kernel(const float* __restrict__ partials,
                                                        float* __restrict__ stats) {
  __shared__ float ssum[256], ssq[256];
  const int b = blockIdx.x, t = threadIdx.x;
  float s = 0.f, q2 = 0.f;
  for (int i = t; i < 1024; i += 256) {            // 1024 tiles per batch
    s  += partials[2 * ((size_t)b * 1024 + i)];
    q2 += partials[2 * ((size_t)b * 1024 + i) + 1];
  }
  ssum[t] = s; ssq[t] = q2; __syncthreads();
  for (int off = 128; off > 0; off >>= 1) {
    if (t < off) { ssum[t] += ssum[t + off]; ssq[t] += ssq[t + off]; }
    __syncthreads();
  }
  if (t == 0) {
    const float n = (float)SS * (float)MM;
    const float mu = ssum[0] / n;
    const float var = ssq[0] / n - mu * mu;
    stats[2 * b] = mu;
    stats[2 * b + 1] = rsqrtf(var + 1e-5f);
  }
}

__global__ __launch_bounds__(256) void ln_apply_kernel(const float* __restrict__ stats,
                                                       const float* __restrict__ lnw,
                                                       const float* __restrict__ lnb,
                                                       float* __restrict__ out) {
  const size_t i = (size_t)blockIdx.x * 256 + threadIdx.x;  // NB*SS*MM
  const int b = (int)(i >> 20);
  const size_t sm = i & ((1u << 20) - 1);
  out[i] = (out[i] - stats[2 * b]) * stats[2 * b + 1] * lnw[sm] + lnb[sm];
}

// ---------------------------------------------------------------------------
extern "C" void kernel_launch(void* const* d_in, const int* in_sizes, int n_in,
                              void* d_out, int out_size, void* d_ws, size_t ws_size,
                              hipStream_t stream) {
  const float*         x    = (const float*)d_in[0];
  const unsigned char* msk  = (const unsigned char*)d_in[1];  // jnp bool = 1 byte
  const float*         w_qk = (const float*)d_in[2];
  const float*         b_qk = (const float*)d_in[3];
  const float*         w_v  = (const float*)d_in[4];
  const float*         b_v  = (const float*)d_in[5];
  const float*         w_o  = (const float*)d_in[6];
  const float*         b_o  = (const float*)d_in[7];
  const float*         ln_w = (const float*)d_in[8];
  const float*         ln_b = (const float*)d_in[9];
  float*               out  = (float*)d_out;

  char* ws = (char*)d_ws;
  size_t off = 0;
  auto carve = [&](size_t bytes) { void* p = ws + off; off += (bytes + 255) & ~(size_t)255; return p; };
  bf16_t* xbf   = (bf16_t*)carve((size_t)NB * SS * MM * 2);       // 16 MB
  bf16_t* wqkvt = (bf16_t*)carve((size_t)NQKV * MM * 2);          //  6 MB
  bf16_t* wot   = (bf16_t*)carve((size_t)MM * MM * 2);            //  2 MB
  bf16_t* qbuf  = (bf16_t*)carve((size_t)HH * NB * SS * KDIM * 2);// 16 MB
  bf16_t* kbuf  = (bf16_t*)carve((size_t)HH * NB * SS * KDIM * 2);// 16 MB
  bf16_t* vbuf  = (bf16_t*)carve((size_t)HH * NB * VDIM * SS * 2);// 16 MB
  bf16_t* obf   = (bf16_t*)carve((size_t)NB * SS * MM * 2);       // 16 MB
  float*  part  = (float*) carve((size_t)8192 * 2 * 4);           // 64 KB
  float*  stats = (float*) carve((size_t)NB * 2 * 4);
  (void)n_in; (void)in_sizes; (void)out_size; (void)ws_size;

  // staging
  cvt_x_kernel     <<<(NB * SS * MM) / 256, 256, 0, stream>>>(x, xbf);
  build_wqkv_kernel<<<(NQKV * MM) / 256, 256, 0, stream>>>(w_qk, w_v, wqkvt);
  build_wo_kernel  <<<(MM * MM) / 256, 256, 0, stream>>>(w_o, wot);
  // fused QKV projection: 512 row-tiles * 48 col-groups / 8 waves per block
  qkv_kernel <<<(512 * 48) / 8, 256, 0, stream>>>(xbf, wqkvt, b_qk, b_v, qbuf, kbuf, vbuf);
  // flash attention: HH*NB*(SS/16) = 8192 wave tasks
  attn_kernel<<<8192 / 8, 256, 0, stream>>>(qbuf, kbuf, vbuf, msk, obf);
  // output projection + residual (res staged in d_out) + LN partial sums
  oproj_kernel<<<(512 * 16) / 8, 256, 0, stream>>>(obf, wot, b_o, x, out, part);
  // per-batch mean / rstd, then elementwise LayerNorm in-place on d_out
  ln_reduce_kernel<<<NB, 256, 0, stream>>>(part, stats);
  ln_apply_kernel <<<(NB * SS * MM) / 256, 256, 0, stream>>>(stats, ln_w, ln_b, out);
}
